// Megnet_EdgeModel_42606075576610
// MI455X (gfx1250) — matrix operand order repaired and checked by reading, not compiled, not run on hardware
//
#include <hip/hip_runtime.h>
#include <hip/hip_bf16.h>
#include <stdint.h>

typedef __attribute__((ext_vector_type(16))) _Float16 v16h;
typedef __attribute__((ext_vector_type(8)))  float    v8f;
typedef __attribute__((ext_vector_type(4)))  float    v4f;

#define D_FEAT 64
#define HDIM   64
#define WAVES_PER_BLOCK 4
#define TILES_PER_WAVE  8
#define TILES_PER_BLOCK (WAVES_PER_BLOCK * TILES_PER_WAVE)

// D = A(16x32 f16) x B(32x16 f16) + C(16x16 f32), wave32.
// A layout: lane<16 -> row m=lane, V0..3=K{hb..hb+7}, V4..7=K{hb+16..hb+23}, hb=0;
//           lane>=16 -> same row m=lane-16, hb=8.
// B layout: lane -> col n=lane&15; V0..3=K{hb..hb+7}, V4..7=K{hb+16..hb+23} (2/dword), hb as in A.
// C/D layout: lane -> col n=lane&15; VGPR j -> row M = 8*(lane>>4) + j.
__global__ __launch_bounds__(128) void megnet_edge_mlp(
    const float* __restrict__ srcF, const float* __restrict__ dstF,
    const float* __restrict__ eaF,  const float* __restrict__ uF,
    const int*   __restrict__ batch,
    const float* __restrict__ W1, const float* __restrict__ B1,
    const float* __restrict__ W2, const float* __restrict__ B2,
    float* __restrict__ out, int nEdges, int nTiles)
{
    // Weights pre-swizzled into WMMA B-fragment order: [kstep][ntile][lane][8 dwords]
    __shared__ __align__(32) uint32_t w1f[8][4][32][8];                  // 32 KB
    __shared__ __align__(32) uint32_t w2f[2][4][32][8];                  //  8 KB
    __shared__ __align__(32) uint32_t hscr[WAVES_PER_BLOCK][2][32][8];   //  8 KB

    const int tid = threadIdx.x;

    // ---- Convert W1 (256x64 f32, row-major) -> f16 B-fragments in LDS ----
    for (int d = tid; d < 8 * 4 * 32 * 8; d += 128) {
        int i = d & 7, lane = (d >> 3) & 31, n = (d >> 8) & 3, kk = d >> 10;
        int k = kk * 32 + ((lane >> 4) << 3) + 2 * i + ((i >> 2) << 3);   // B: table order, hb=8*half, V4..7 -> +16
        int c = n * 16 + (lane & 15);
        union { _Float16 h[2]; uint32_t q; } p;
        p.h[0] = (_Float16)W1[(long)k * HDIM + c];
        p.h[1] = (_Float16)W1[(long)(k + 1) * HDIM + c];
        w1f[kk][n][lane][i] = p.q;
    }
    // ---- Convert W2 (64x64) likewise ----
    for (int d = tid; d < 2 * 4 * 32 * 8; d += 128) {
        int i = d & 7, lane = (d >> 3) & 31, n = (d >> 8) & 3, kk = d >> 10;
        int k = kk * 32 + ((lane >> 4) << 3) + 2 * i + ((i >> 2) << 3);
        int c = n * 16 + (lane & 15);
        union { _Float16 h[2]; uint32_t q; } p;
        p.h[0] = (_Float16)W2[(long)k * HDIM + c];
        p.h[1] = (_Float16)W2[(long)(k + 1) * HDIM + c];
        w2f[kk][n][lane][i] = p.q;
    }
    __syncthreads();

    const int lane  = tid & 31;
    const int wave  = tid >> 5;
    const int m     = lane & 15;     // row within 16-edge tile (A) / column (B, C/D)
    const int halfH = lane >> 4;
    const int hb    = halfH * 8;     // A-layout K sub-base for this lane half

    for (int it = 0; it < TILES_PER_WAVE; ++it) {
        int t = blockIdx.x * TILES_PER_BLOCK + it * WAVES_PER_BLOCK + wave;
        if (t >= nTiles) break;                      // uniform per wave

        long e  = (long)t * 16 + m;
        long ec = e < nEdges ? e : (long)nEdges - 1; // clamp for tail loads
        const float* p0 = srcF + ec * D_FEAT;
        const float* p1 = dstF + ec * D_FEAT;
        const float* p2 = eaF  + ec * D_FEAT;
        const float* p3 = uF   + (long)batch[ec] * D_FEAT;

        v8f acc[4];
#pragma unroll
        for (int n = 0; n < 4; ++n) acc[n] = (v8f){0.f,0.f,0.f,0.f,0.f,0.f,0.f,0.f};

        // ---- GEMM1: comb[16x256] @ W1[256x64], K in 8 steps of 32 ----
#pragma unroll
        for (int kk = 0; kk < 8; ++kk) {
            const float* ps = (kk < 2) ? p0 : (kk < 4) ? p1 : (kk < 6) ? p2 : p3;
            int c0 = ((kk & 1) << 5) + hb;           // comb col = kk*32 + hb (+0..7,+16..23)
            v4f fa = *(const v4f*)(ps + c0);
            v4f fb = *(const v4f*)(ps + c0 + 4);
            v4f fc = *(const v4f*)(ps + c0 + 16);
            v4f fd = *(const v4f*)(ps + c0 + 20);
            v16h a;
#pragma unroll
            for (int j = 0; j < 4; ++j) {
                a[j]      = (_Float16)fa[j];
                a[4 + j]  = (_Float16)fb[j];
                a[8 + j]  = (_Float16)fc[j];
                a[12 + j] = (_Float16)fd[j];
            }
#pragma unroll
            for (int n = 0; n < 4; ++n) {
                v16h b = *(const v16h*)&w1f[kk][n][lane][0];
                acc[n] = __builtin_amdgcn_wmma_f32_16x16x32_f16(
                    false, a, false, b, (short)0, acc[n], false, false);
            }
        }

        // ---- bias + ReLU; scatter h (C/D layout) into A-layout LDS scratch ----
#pragma unroll
        for (int n = 0; n < 4; ++n) {
            int c    = n * 16 + m;                   // h column = K index for GEMM2
            float bv = B1[c];
            int kk2  = c >> 5;
            int kl   = c & 31;
            int hsel = (kl >> 3) & 1;                // which lane-half holds this K
            int i2   = ((kl & 16) ? 4 : 0) + ((kl & 7) >> 1);
            int pos  = kl & 1;
#pragma unroll
            for (int j = 0; j < 8; ++j) {
                float hv = acc[n][j] + bv;
                hv = hv > 0.f ? hv : 0.f;
                int lane_a = (halfH * 8 + j) + 16 * hsel;   // row M -> A lane
                _Float16* q = (_Float16*)&hscr[wave][kk2][lane_a][i2];
                q[pos] = (_Float16)hv;
            }
        }
        __asm volatile("s_wait_dscnt 0" ::: "memory");

        // ---- GEMM2: h[16x64] @ W2[64x64], K in 2 steps ----
        v16h a0 = *(const v16h*)&hscr[wave][0][lane][0];
        v16h a1 = *(const v16h*)&hscr[wave][1][lane][0];
#pragma unroll
        for (int n = 0; n < 4; ++n) {
            v8f o = (v8f){0.f,0.f,0.f,0.f,0.f,0.f,0.f,0.f};
            v16h w0 = *(const v16h*)&w2f[0][n][lane][0];
            o = __builtin_amdgcn_wmma_f32_16x16x32_f16(false, a0, false, w0, (short)0, o, false, false);
            v16h w1q = *(const v16h*)&w2f[1][n][lane][0];
            o = __builtin_amdgcn_wmma_f32_16x16x32_f16(false, a1, false, w1q, (short)0, o, false, false);

            int c    = n * 16 + m;
            float bv = B2[c];
#pragma unroll
            for (int j = 0; j < 8; ++j) {
                float ov = o[j] + bv;
                ov = ov > 0.f ? ov : 0.f;
                long eo = (long)t * 16 + halfH * 8 + j;
                if (eo < nEdges) out[eo * HDIM + c] = ov;
            }
        }
    }
}

extern "C" void kernel_launch(void* const* d_in, const int* in_sizes, int n_in,
                              void* d_out, int out_size, void* d_ws, size_t ws_size,
                              hipStream_t stream) {
    const float* srcF  = (const float*)d_in[0];
    const float* dstF  = (const float*)d_in[1];
    const float* eaF   = (const float*)d_in[2];
    const float* uF    = (const float*)d_in[3];
    const int*   batch = (const int*)  d_in[4];
    const float* W1    = (const float*)d_in[5];
    const float* B1    = (const float*)d_in[6];
    const float* W2    = (const float*)d_in[7];
    const float* B2    = (const float*)d_in[8];
    float* out = (float*)d_out;

    int nEdges = in_sizes[0] / D_FEAT;                 // 500000
    int nTiles = (nEdges + 15) / 16;                   // 31250
    int grid   = (nTiles + TILES_PER_BLOCK - 1) / TILES_PER_BLOCK;

    megnet_edge_mlp<<<grid, WAVES_PER_BLOCK * 32, 0, stream>>>(
        srcF, dstF, eaF, uF, batch, W1, B1, W2, B2, out, nEdges, nTiles);
}
